// MultiHeadAttentionBlock_25563645345916
// MI455X (gfx1250) — matrix operand order repaired and checked by reading, not compiled, not run on hardware
//
#include <hip/hip_runtime.h>
#include <hip/hip_bf16.h>

#define B_DIM 2
#define S_LEN 2048
#define D_DIM 1024
#define NH    16
#define DKH   64
#define MTOT  (B_DIM * S_LEN)   // 4096

typedef float  v8f  __attribute__((ext_vector_type(8)));
typedef __bf16 v16bf __attribute__((ext_vector_type(16)));

union Frag16 {
    v16bf v;
    uint4 u4[2];
    unsigned short s[16];
};

__device__ __forceinline__ unsigned short f32_to_bf16(float f) {
    union { float f; unsigned int u; } x{f};
    unsigned int r = x.u + 0x7FFFu + ((x.u >> 16) & 1u);   // round-to-nearest-even
    return (unsigned short)(r >> 16);
}

// ---------------------------------------------------------------------------
// CDNA5 async global->LDS (ASYNCcnt path). The instruction offset is added to
// BOTH the LDS and the global address (ISA 08_async_tensor §4.4), so one
// (lds, voff) pair covers a 32B chunk with offset:0 / offset:16.
// ---------------------------------------------------------------------------
__device__ __forceinline__ void async_ld_b128(unsigned lds, unsigned voff, const void* sbase) {
    asm volatile("global_load_async_to_lds_b128 %0, %1, %2"
                 :: "v"(lds), "v"(voff), "s"(sbase) : "memory");
}
__device__ __forceinline__ void async_ld_b128_o16(unsigned lds, unsigned voff, const void* sbase) {
    asm volatile("global_load_async_to_lds_b128 %0, %1, %2 offset:16"
                 :: "v"(lds), "v"(voff), "s"(sbase) : "memory");
}
__device__ __forceinline__ void wait_async0() {
    asm volatile("s_wait_asynccnt 0x0" ::: "memory");
}
__device__ __forceinline__ unsigned lds_off(const void* p) {
    return (unsigned)(size_t)p;   // flat LDS address: addr[31:0] == LDS offset
}

// ---------------------------------------------------------------------------
// fp32 -> bf16 conversion, 8 elements per thread, b128 in / b128 out
// ---------------------------------------------------------------------------
__global__ __launch_bounds__(256) void cvt_f32_bf16(const float* __restrict__ in,
                                                    unsigned short* __restrict__ out,
                                                    int n) {
    int i = (blockIdx.x * 256 + threadIdx.x) * 8;
    if (i + 8 > n) return;
    float4 a = *(const float4*)(in + i);
    float4 b = *(const float4*)(in + i + 4);
    union { unsigned short s[8]; uint4 u; } r;
    r.s[0] = f32_to_bf16(a.x); r.s[1] = f32_to_bf16(a.y);
    r.s[2] = f32_to_bf16(a.z); r.s[3] = f32_to_bf16(a.w);
    r.s[4] = f32_to_bf16(b.x); r.s[5] = f32_to_bf16(b.y);
    r.s[6] = f32_to_bf16(b.z); r.s[7] = f32_to_bf16(b.w);
    *(uint4*)(out + i) = r.u;
}

// ---------------------------------------------------------------------------
// Y[M,N] = X[M,K] @ W[N,K]^T + bias[N]
// 256 threads = 8 waves, block tile 128x128, wave tile 64x32 (4x2 WMMA accs).
// Double-buffered 128x32 bf16 LDS panels filled by async global->LDS; next
// panel streams in while WMMAs run on the current one. One barrier per step.
// ---------------------------------------------------------------------------
__global__ __launch_bounds__(256) void gemm_xwT_bias(
        const unsigned short* __restrict__ X,
        const unsigned short* __restrict__ W,
        const float* __restrict__ bias,
        unsigned short* __restrict__ Ybf,
        float* __restrict__ Yf,
        int M, int N, int K, int store_f32) {
    __shared__ __attribute__((aligned(16))) unsigned short As[2][128 * 32];
    __shared__ __attribute__((aligned(16))) unsigned short Bs[2][128 * 32];

    const int tid   = threadIdx.x;
    const int lane  = tid & 31;
    const int wave  = tid >> 5;
    const int l15   = lane & 15;
    const bool hi   = lane >= 16;
    const int waveM = wave & 1;       // 0..1 -> 64-row groups
    const int waveN = wave >> 1;      // 0..3 -> 32-col groups
    const int m0 = blockIdx.y * 128;
    const int n0 = blockIdx.x * 128;

    v8f acc[4][2];
#pragma unroll
    for (int rb = 0; rb < 4; ++rb)
#pragma unroll
        for (int cb = 0; cb < 2; ++cb) acc[rb][cb] = v8f{};

    const int lr = tid >> 1;          // 0..127 : panel row loaded by this thread
    const int lk = (tid & 1) * 16;    // 0 / 16 : k-offset within 32-wide panel
    const int klo = hi ? 8 : 0;       // A-fragment K split (lo/hi half-wave)

    const unsigned ldsA0 = lds_off(&As[0][lr * 32 + lk]);
    const unsigned ldsB0 = lds_off(&Bs[0][lr * 32 + lk]);
    const unsigned bufB  = 128 * 32 * 2;             // panel stride in bytes
    const unsigned voff  = (unsigned)((lr * K + lk) * 2);
    const char* Xb = (const char*)(X + (size_t)m0 * K);
    const char* Wb = (const char*)(W + (size_t)n0 * K);

    // prologue: stream panel 0 into buffer 0
    async_ld_b128    (ldsA0, voff, Xb);
    async_ld_b128_o16(ldsA0, voff, Xb);
    async_ld_b128    (ldsB0, voff, Wb);
    async_ld_b128_o16(ldsB0, voff, Wb);

    const int nsteps = K / 32;
    for (int it = 0; it < nsteps; ++it) {
        const int buf = it & 1;
        wait_async0();        // own transfers into `buf` complete
        __syncthreads();      // everyone's transfers complete, prior reads done
        if (it + 1 < nsteps) {
            const unsigned nb = (unsigned)(buf ^ 1) * bufB;
            const char* Xn = Xb + (size_t)(it + 1) * 64;   // 32 elems * 2B
            const char* Wn = Wb + (size_t)(it + 1) * 64;
            async_ld_b128    (ldsA0 + nb, voff, Xn);
            async_ld_b128_o16(ldsA0 + nb, voff, Xn);
            async_ld_b128    (ldsB0 + nb, voff, Wn);
            async_ld_b128_o16(ldsB0 + nb, voff, Wn);
        }

        v16bf af[4], bfr[2];
#pragma unroll
        for (int rb = 0; rb < 4; ++rb) {
            const unsigned short* p = &As[buf][(waveM * 64 + rb * 16 + l15) * 32 + klo];
            Frag16 f; f.u4[0] = *(const uint4*)p; f.u4[1] = *(const uint4*)(p + 16);
            af[rb] = f.v;
        }
#pragma unroll
        for (int cb = 0; cb < 2; ++cb) {
            const unsigned short* p = &Bs[buf][(waveN * 32 + cb * 16 + l15) * 32 + klo];
            Frag16 f; f.u4[0] = *(const uint4*)p; f.u4[1] = *(const uint4*)(p + 16);
            bfr[cb] = f.v;
        }
#pragma unroll
        for (int rb = 0; rb < 4; ++rb)
#pragma unroll
            for (int cb = 0; cb < 2; ++cb)
                acc[rb][cb] = __builtin_amdgcn_wmma_f32_16x16x32_bf16(
                    false, af[rb], false, bfr[cb], (short)0, acc[rb][cb], false, false);
    }

#pragma unroll
    for (int cb = 0; cb < 2; ++cb) {
        const int n = n0 + waveN * 32 + cb * 16 + l15;
        const float bv = bias[n];
#pragma unroll
        for (int rb = 0; rb < 4; ++rb) {
#pragma unroll
            for (int r = 0; r < 8; ++r) {
                const int m = m0 + waveM * 64 + rb * 16 + r + (hi ? 8 : 0);
                const float val = acc[rb][cb][r] + bv;
                if (store_f32) Yf[(size_t)m * N + n] = val;
                else           Ybf[(size_t)m * N + n] = f32_to_bf16(val);
            }
        }
    }
}

// ---------------------------------------------------------------------------
// Causal flash attention over one (batch, head) per blockIdx.y.
// Block = 128 q rows (8 waves x 16 rows), KV tiles of 64 staged in LDS:
//   Ks row-major [kv][dk]  (async global->LDS) -> contiguous score B-frags
//   Vt transposed [dk][kv] (VGPR path)         -> contiguous P@V B-frags
// Per-wave P staging converts score C-layout -> WMMA A-layout.
// ---------------------------------------------------------------------------
__global__ __launch_bounds__(256) void flash_attn_bf16(
        const unsigned short* __restrict__ Qp,
        const unsigned short* __restrict__ Kp,
        const unsigned short* __restrict__ Vp,
        unsigned short* __restrict__ Ctx) {
    __shared__ __attribute__((aligned(16))) unsigned short Ks[64 * 64];
    __shared__ __attribute__((aligned(16))) unsigned short Vt[64 * 64];
    __shared__ __attribute__((aligned(16))) unsigned short Ps[8][16 * 64];

    const int tid  = threadIdx.x;
    const int wave = tid >> 5;
    const int lane = tid & 31;
    const int l15  = lane & 15;
    const bool hi  = lane >= 16;
    const int klo  = hi ? 8 : 0;

    const int qb = blockIdx.x * 128;
    const int bh = blockIdx.y;
    const int b  = bh / NH;
    const int h  = bh % NH;

    const int qrow_base = qb + wave * 16;
    const size_t rowQ = (size_t)(b * S_LEN + qrow_base + l15) * D_DIM + h * DKH;

    // persistent Q fragments: 16 rows x 64 dk = 2 chunks of K=32
    v16bf qfrag[2];
#pragma unroll
    for (int c = 0; c < 2; ++c) {
        const unsigned short* p = Qp + rowQ + c * 32 + klo;
        Frag16 f; f.u4[0] = *(const uint4*)p; f.u4[1] = *(const uint4*)(p + 16);
        qfrag[c] = f.v;
    }

    float mrow[8], lrow[8];
    v8f accO[4];
#pragma unroll
    for (int r = 0; r < 8; ++r) { mrow[r] = -3.0e38f; lrow[r] = 0.0f; }
#pragma unroll
    for (int nb = 0; nb < 4; ++nb) accO[nb] = v8f{};

    const int r_  = tid >> 2;         // 0..63 kv row staged by this thread
    const int cg  = (tid & 3) * 16;   // dk column group
    const unsigned ldsK = lds_off(&Ks[r_ * 64 + cg]);
    const unsigned kvo  = (unsigned)((r_ * D_DIM + cg) * 2);

    const int num_jt = qb / 64 + 2;   // cover causal range of all 128 rows
    for (int jt = 0; jt < num_jt; ++jt) {
        const int j0 = jt * 64;
        // ---- cooperative KV staging ----
        {
            const size_t kvrow = (size_t)(b * S_LEN + j0 + r_) * D_DIM + h * DKH + cg;
            Frag16 fv;
            fv.u4[0] = *(const uint4*)(Vp + kvrow);
            fv.u4[1] = *(const uint4*)(Vp + kvrow + 8);
            __syncthreads();                 // prior tile fully consumed
            const char* kbase = (const char*)(Kp + (size_t)(b * S_LEN + j0) * D_DIM + h * DKH);
            async_ld_b128    (ldsK, kvo, kbase);     // K tile via ASYNCcnt path
            async_ld_b128_o16(ldsK, kvo, kbase);
#pragma unroll
            for (int i = 0; i < 16; ++i) Vt[(cg + i) * 64 + r_] = fv.s[i];
            wait_async0();
            __syncthreads();
        }

        // ---- scores: 16 q rows x 64 kv, 4 col-blocks x 2 K-chunks ----
        v8f sc[4];
#pragma unroll
        for (int cb = 0; cb < 4; ++cb) {
            v8f a = v8f{};
#pragma unroll
            for (int c = 0; c < 2; ++c) {
                const unsigned short* p = &Ks[(cb * 16 + l15) * 64 + c * 32 + klo];
                Frag16 f; f.u4[0] = *(const uint4*)p; f.u4[1] = *(const uint4*)(p + 16);
                a = __builtin_amdgcn_wmma_f32_16x16x32_bf16(
                        false, qfrag[c], false, f.v, (short)0, a, false, false);
            }
            sc[cb] = a;
        }

        // ---- scale + causal mask (C layout: row = r + hi*8, col = l15) ----
#pragma unroll
        for (int cb = 0; cb < 4; ++cb) {
            const int kv = j0 + cb * 16 + l15;
#pragma unroll
            for (int r = 0; r < 8; ++r) {
                const int qr = qrow_base + r + (hi ? 8 : 0);
                const float s = sc[cb][r] * 0.125f;   // 1/sqrt(64)
                sc[cb][r] = (kv <= qr) ? s : -1.0e9f;
            }
        }

        // ---- online softmax: row reductions over 16-lane halves ----
        float rmax[8], alpha[8], rsum[8];
#pragma unroll
        for (int r = 0; r < 8; ++r) {
            float m = fmaxf(fmaxf(sc[0][r], sc[1][r]), fmaxf(sc[2][r], sc[3][r]));
#pragma unroll
            for (int msk = 1; msk < 16; msk <<= 1)
                m = fmaxf(m, __shfl_xor(m, msk, 32));
            rmax[r] = m;
        }
#pragma unroll
        for (int r = 0; r < 8; ++r) {
            const float mn = fmaxf(mrow[r], rmax[r]);
            alpha[r] = __expf(mrow[r] - mn);
            mrow[r] = mn;
            rsum[r] = 0.0f;
        }
#pragma unroll
        for (int cb = 0; cb < 4; ++cb) {
#pragma unroll
            for (int r = 0; r < 8; ++r) {
                const float p = __expf(sc[cb][r] - mrow[r]);
                rsum[r] += p;
                Ps[wave][(r + (hi ? 8 : 0)) * 64 + cb * 16 + l15] = f32_to_bf16(p);
            }
        }
#pragma unroll
        for (int r = 0; r < 8; ++r) {
            float s = rsum[r];
#pragma unroll
            for (int msk = 1; msk < 16; msk <<= 1)
                s += __shfl_xor(s, msk, 32);
            lrow[r] = lrow[r] * alpha[r] + s;
        }
#pragma unroll
        for (int nb = 0; nb < 4; ++nb)
#pragma unroll
            for (int r = 0; r < 8; ++r) accO[nb][r] *= alpha[r];

        // ---- P @ V : reload P in A-layout, V^T fragments from LDS ----
        v16bf pfrag[2];
#pragma unroll
        for (int c = 0; c < 2; ++c) {
            const unsigned short* p = &Ps[wave][l15 * 64 + c * 32 + klo];
            Frag16 f; f.u4[0] = *(const uint4*)p; f.u4[1] = *(const uint4*)(p + 16);
            pfrag[c] = f.v;
        }
#pragma unroll
        for (int nb = 0; nb < 4; ++nb) {
#pragma unroll
            for (int c = 0; c < 2; ++c) {
                const unsigned short* p = &Vt[(nb * 16 + l15) * 64 + c * 32 + klo];
                Frag16 f; f.u4[0] = *(const uint4*)p; f.u4[1] = *(const uint4*)(p + 16);
                accO[nb] = __builtin_amdgcn_wmma_f32_16x16x32_bf16(
                        false, pfrag[c], false, f.v, (short)0, accO[nb], false, false);
            }
        }
    }

    // ---- normalize and store context (bf16, merged-head layout) ----
    float linv[8];
#pragma unroll
    for (int r = 0; r < 8; ++r) linv[r] = 1.0f / lrow[r];
#pragma unroll
    for (int nb = 0; nb < 4; ++nb) {
#pragma unroll
        for (int r = 0; r < 8; ++r) {
            const int row = qrow_base + r + (hi ? 8 : 0);
            const int dk  = nb * 16 + l15;
            Ctx[(size_t)(b * S_LEN + row) * D_DIM + h * DKH + dk] =
                f32_to_bf16(accO[nb][r] * linv[r]);
        }
    }
}

// ---------------------------------------------------------------------------
// Host-side launch sequence
// ---------------------------------------------------------------------------
extern "C" void kernel_launch(void* const* d_in, const int* in_sizes, int n_in,
                              void* d_out, int out_size, void* d_ws, size_t ws_size,
                              hipStream_t stream) {
    const float* q  = (const float*)d_in[0];
    const float* k  = (const float*)d_in[1];
    const float* v  = (const float*)d_in[2];
    // d_in[3] = causal mask, computed analytically in-kernel
    const float* Wq = (const float*)d_in[4];
    const float* bq = (const float*)d_in[5];
    const float* Wk = (const float*)d_in[6];
    const float* bk = (const float*)d_in[7];
    const float* Wv = (const float*)d_in[8];
    const float* bv = (const float*)d_in[9];
    const float* Wo = (const float*)d_in[10];
    const float* bo = (const float*)d_in[11];
    float* out = (float*)d_out;

    char* ws = (char*)d_ws;
    const size_t MB = 1024u * 1024u;
    unsigned short* qb  = (unsigned short*)(ws + 0 * MB);   // 8 MB each
    unsigned short* kb  = (unsigned short*)(ws + 8 * MB);
    unsigned short* vb  = (unsigned short*)(ws + 16 * MB);
    unsigned short* Qp  = (unsigned short*)(ws + 24 * MB);
    unsigned short* Kp  = (unsigned short*)(ws + 32 * MB);
    unsigned short* Vp  = (unsigned short*)(ws + 40 * MB);
    unsigned short* Ctx = (unsigned short*)(ws + 48 * MB);
    unsigned short* Wqb = (unsigned short*)(ws + 56 * MB);  // 2 MB each
    unsigned short* Wkb = (unsigned short*)(ws + 58 * MB);
    unsigned short* Wvb = (unsigned short*)(ws + 60 * MB);
    unsigned short* Wob = (unsigned short*)(ws + 62 * MB);

    const int nbig = MTOT * D_DIM;   // 4,194,304
    const int nw   = D_DIM * D_DIM;  // 1,048,576
    cvt_f32_bf16<<<nbig / 2048, 256, 0, stream>>>(q,  qb,  nbig);
    cvt_f32_bf16<<<nbig / 2048, 256, 0, stream>>>(k,  kb,  nbig);
    cvt_f32_bf16<<<nbig / 2048, 256, 0, stream>>>(v,  vb,  nbig);
    cvt_f32_bf16<<<nw   / 2048, 256, 0, stream>>>(Wq, Wqb, nw);
    cvt_f32_bf16<<<nw   / 2048, 256, 0, stream>>>(Wk, Wkb, nw);
    cvt_f32_bf16<<<nw   / 2048, 256, 0, stream>>>(Wv, Wvb, nw);
    cvt_f32_bf16<<<nw   / 2048, 256, 0, stream>>>(Wo, Wob, nw);

    dim3 gg(D_DIM / 128, MTOT / 128);   // (8, 32)
    gemm_xwT_bias<<<gg, 256, 0, stream>>>(qb, Wqb, bq, Qp, nullptr, MTOT, D_DIM, D_DIM, 0);
    gemm_xwT_bias<<<gg, 256, 0, stream>>>(kb, Wkb, bk, Kp, nullptr, MTOT, D_DIM, D_DIM, 0);
    gemm_xwT_bias<<<gg, 256, 0, stream>>>(vb, Wvb, bv, Vp, nullptr, MTOT, D_DIM, D_DIM, 0);

    dim3 fg(S_LEN / 128, B_DIM * NH);   // (16, 32)
    flash_attn_bf16<<<fg, 256, 0, stream>>>(Qp, Kp, Vp, Ctx);

    gemm_xwT_bias<<<gg, 256, 0, stream>>>(Ctx, Wob, bo, nullptr, out, MTOT, D_DIM, D_DIM, 1);
}
